// EuclideanTransformer_13889924235380
// MI455X (gfx1250) — matrix operand order repaired and checked by reading, not compiled, not run on hardware
//
#include <hip/hip_runtime.h>
#include <hip/hip_bf16.h>

// ---------------------------------------------------------------------------
// Euclidean graph transformer layer for MI455X (gfx1250, wave32, WMMA).
// Filter MLPs (~21 of 22 GFLOP) run on v_wmma_f32_16x16x32_bf16 with f32
// accumulation; everything else is exact f32 VALU.
// ---------------------------------------------------------------------------

#define N_NODES_ 20000
#define N_EDGES_ 200000
#define F_  132     // invariant feature width
#define EVD_ 15     // equivariant feature width
#define NRB_ 32     // radial basis count
#define NP_ 144     // padded N dim for GEMMs (9 tiles of 16)
#define KP_ 192     // padded K dim for GEMM2 (6 blocks of 32): 132 r + 33 e + 27 pad

typedef __attribute__((ext_vector_type(16))) __bf16 v16bf;
typedef __attribute__((ext_vector_type(8)))  float  v8f;

union V16 { v16bf bf; unsigned short us[16]; };

__device__ __forceinline__ unsigned short f2bf(float f) {
    unsigned u = __float_as_uint(f);
    u += 0x7FFFu + ((u >> 16) & 1u);   // round-to-nearest-even
    return (unsigned short)(u >> 16);
}

// silu with fast hardware reciprocal (v_rcp_f32) instead of IEEE divide
__device__ __forceinline__ float silu_f(float x) {
    return x * __builtin_amdgcn_rcpf(1.0f + __expf(-x));
}

// ---------------------------------------------------------------------------
// K0: pre-pack weights: bf16-convert + pad W1 (32x132 -> 32x144) and the
// concatenated GEMM2 weight [W2(132x132); We2(33x132); 0] -> 192x144,
// and fuse biases b2+be2.
// ---------------------------------------------------------------------------
__global__ void et_prep_kernel(
    const float* __restrict__ fiW1, const float* __restrict__ fiW2,
    const float* __restrict__ fiEvW2, const float* __restrict__ fib2,
    const float* __restrict__ fiEvb2,
    const float* __restrict__ feW1, const float* __restrict__ feW2,
    const float* __restrict__ feEvW2, const float* __restrict__ feb2,
    const float* __restrict__ feEvb2,
    unsigned short* __restrict__ fiW1p, unsigned short* __restrict__ fiWcat,
    unsigned short* __restrict__ feW1p, unsigned short* __restrict__ feWcat,
    float* __restrict__ bcat_fi, float* __restrict__ bcat_fe)
{
    int t = blockIdx.x * blockDim.x + threadIdx.x;
    const int A = NRB_ * NP_;   // 4608
    const int B = KP_ * NP_;    // 27648
    if (t < A) {
        int k = t / NP_, j = t % NP_;
        fiW1p[t] = (j < F_) ? f2bf(fiW1[k * F_ + j]) : (unsigned short)0;
        return;
    }
    t -= A;
    if (t < A) {
        int k = t / NP_, j = t % NP_;
        feW1p[t] = (j < F_) ? f2bf(feW1[k * F_ + j]) : (unsigned short)0;
        return;
    }
    t -= A;
    if (t < B) {
        int k = t / NP_, j = t % NP_;
        float v = 0.f;
        if (j < F_) {
            if (k < F_) v = fiW2[k * F_ + j];
            else if (k < F_ + 33) v = fiEvW2[(k - F_) * F_ + j];
        }
        fiWcat[t] = f2bf(v);
        return;
    }
    t -= B;
    if (t < B) {
        int k = t / NP_, j = t % NP_;
        float v = 0.f;
        if (j < F_) {
            if (k < F_) v = feW2[k * F_ + j];
            else if (k < F_ + 33) v = feEvW2[(k - F_) * F_ + j];
        }
        feWcat[t] = f2bf(v);
        return;
    }
    t -= B;
    if (t < F_) { bcat_fi[t] = fib2[t] + fiEvb2[t]; return; }
    t -= F_;
    if (t < F_) { bcat_fe[t] = feb2[t] + feEvb2[t]; return; }
}

// ---------------------------------------------------------------------------
// K1: node q/k/v projections (per-head 33x33 for inv, 44x44 for ev; note
// q_ev/k_ev project inv_features per the reference) + zero scatter buffers.
// ---------------------------------------------------------------------------
__global__ void et_node_proj_kernel(
    const float* __restrict__ x,
    const float* __restrict__ Wq, const float* __restrict__ Wk,
    const float* __restrict__ Wv,
    const float* __restrict__ Wqe, const float* __restrict__ Wke,
    float* __restrict__ q_inv, float* __restrict__ k_inv,
    float* __restrict__ v_inv,
    float* __restrict__ q_ev, float* __restrict__ k_ev,
    float* __restrict__ d_inv, float* __restrict__ d_ev)
{
    int t = blockIdx.x * blockDim.x + threadIdx.x;
    if (t >= N_NODES_ * F_) return;
    int n = t / F_, f = t % F_;
    const float* xr = x + (size_t)n * F_;
    {   // inv heads: h = f/33, j = f%33 ; out = sum_i W[h,i,j] * x[n, h*33+i]
        int h = f / 33, j = f % 33;
        const float* wq = Wq + h * 1089 + j;
        const float* wk = Wk + h * 1089 + j;
        const float* wv = Wv + h * 1089 + j;
        const float* xh = xr + h * 33;
        float sq = 0.f, sk = 0.f, sv = 0.f;
        #pragma unroll
        for (int i = 0; i < 33; ++i) {
            float xi = xh[i];
            sq += wq[i * 33] * xi;
            sk += wk[i * 33] * xi;
            sv += wv[i * 33] * xi;
        }
        q_inv[t] = sq; k_inv[t] = sk; v_inv[t] = sv;
    }
    {   // ev heads: h = f/44, j = f%44
        int h = f / 44, j = f % 44;
        const float* wq = Wqe + h * 1936 + j;
        const float* wk = Wke + h * 1936 + j;
        const float* xh = xr + h * 44;
        float sq = 0.f, sk = 0.f;
        #pragma unroll
        for (int i = 0; i < 44; ++i) {
            float xi = xh[i];
            sq += wq[i * 44] * xi;
            sk += wk[i * 44] * xi;
        }
        q_ev[t] = sq; k_ev[t] = sk;
    }
    d_inv[t] = 0.f;
    if (f < EVD_) d_ev[n * EVD_ + f] = 0.f;
}

// ---------------------------------------------------------------------------
// K2: edge kernel. One wave = one 16-edge tile. Filter MLPs via WMMA bf16,
// attention + scatter-add in f32.
// ---------------------------------------------------------------------------
__global__ __launch_bounds__(32) void et_edge_kernel(
    const float* __restrict__ ev,        // (N,15)
    const float* __restrict__ rbf,       // (E,32)
    const int*   __restrict__ senders,
    const int*   __restrict__ receivers,
    const float* __restrict__ sh,        // (E,15)
    const float* __restrict__ cutoffs,   // (E,1)
    const float* __restrict__ fi_b1, const float* __restrict__ fi_evW1,
    const float* __restrict__ fi_evb1,
    const float* __restrict__ fe_b1, const float* __restrict__ fe_evW1,
    const float* __restrict__ fe_evb1,
    const unsigned short* __restrict__ fiW1p,   // (32,144) bf16
    const unsigned short* __restrict__ fiWcat,  // (192,144) bf16
    const unsigned short* __restrict__ feW1p,
    const unsigned short* __restrict__ feWcat,
    const float* __restrict__ bcat_fi, const float* __restrict__ bcat_fe,
    const float* __restrict__ q_inv, const float* __restrict__ k_inv,
    const float* __restrict__ v_inv,
    const float* __restrict__ q_ev, const float* __restrict__ k_ev,
    float* __restrict__ d_inv,           // (N,132) accumulators
    float* __restrict__ d_ev)            // (N,15)  accumulators
{
    __shared__ unsigned short Acat[16][KP_];  // bf16 staging (6144 B)
    __shared__ float fw_fi[16][F_];           // fi filter out (8448 B)
    __shared__ float fw_fe[16][F_];           // fe filter out (8448 B)

    const int lane = threadIdx.x;    // 0..31 (wave32)
    const int m    = lane & 15;      // edge row within tile / D-layout column
    const int hf   = lane >> 4;      // 0/1
    const int e    = blockIdx.x * 16 + m;

    // ---- per-edge scalars (computed by both lanes of a pair; cache-hot) ----
    const int sN = senders[e];
    const int rN = receivers[e];
    const float cut = cutoffs[e];

    float evi0 = 0.f, evi1 = 0.f, evi2 = 0.f;
    {
        const float* evs = ev + (size_t)sN * EVD_;
        const float* evr = ev + (size_t)rN * EVD_;
        #pragma unroll
        for (int i = 0; i < EVD_; ++i) {
            float d = evs[i] - evr[i];
            float p = d * d;
            if (i < 3) evi0 += p; else if (i < 8) evi1 += p; else evi2 += p;
        }
    }

    // ---- A fragment for GEMM1: 16x32 rbf tile, bf16, ISA A-layout ----
    V16 aR;
    {
        const float* rr = rbf + (size_t)e * NRB_;
        #pragma unroll
        for (int i = 0; i < 8; ++i) aR.us[i]     = f2bf(rr[hf * 8 + i]);
        #pragma unroll
        for (int i = 0; i < 8; ++i) aR.us[8 + i] = f2bf(rr[16 + hf * 8 + i]);
    }

    const v8f vzero = {0.f, 0.f, 0.f, 0.f, 0.f, 0.f, 0.f, 0.f};

    auto run_net = [&](const unsigned short* __restrict__ W1p,
                       const unsigned short* __restrict__ Wcat,
                       const float* __restrict__ b1,
                       const float* __restrict__ evW1,
                       const float* __restrict__ evb1,
                       const float* __restrict__ bcat,
                       float (*fwOut)[F_]) {
        // ---- GEMM1: (16x32) x (32x144) : preload 9 B fragments, 9 WMMA ----
        v8f c1[9];
        {
            V16 b1v[9];
            #pragma unroll
            for (int t = 0; t < 9; ++t) {
                const unsigned short* bp = W1p + t * 16 + m;  // lane = N column
                #pragma unroll
                for (int i = 0; i < 8; ++i) {
                    b1v[t].us[i]     = bp[(hf * 8 + i) * NP_];
                    b1v[t].us[8 + i] = bp[(16 + hf * 8 + i) * NP_];
                }
            }
            #pragma unroll
            for (int t = 0; t < 9; ++t) {
                c1[t] = __builtin_amdgcn_wmma_f32_16x16x32_bf16(
                    false, aR.bf, false, b1v[t].bf, (short)0, vzero, false, false);
            }
        }

        __syncthreads();  // previous net's Acat reads done before overwrite

        // ---- bias + silu -> Acat rows [K = 0..131] (D-layout: col = m) ----
        #pragma unroll
        for (int t = 0; t < 9; ++t) {
            int f = t * 16 + m;
            if (f < F_) {
                float bb = b1[f];
                #pragma unroll
                for (int r = 0; r < 8; ++r) {
                    float x = c1[t][r] + bb;
                    Acat[hf * 8 + r][f] = f2bf(silu_f(x));
                }
            }
        }
        // ---- e-path hidden: silu(evi @ We1 + be1) -> Acat K = 132..164 ----
        #pragma unroll
        for (int jj = 0; jj < 17; ++jj) {
            int j = hf * 17 + jj;
            if (j < 33) {
                float x = evb1[j] + evi0 * evW1[j] + evi1 * evW1[33 + j]
                                  + evi2 * evW1[66 + j];
                Acat[m][F_ + j] = f2bf(silu_f(x));
            }
        }
        // ---- zero pad K = 165..191 ----
        {
            int p0 = 165 + hf * 14;
            int p1 = hf ? KP_ : 179;
            #pragma unroll
            for (int j = 0; j < 14; ++j)
                if (p0 + j < p1) Acat[m][p0 + j] = 0;
        }
        __syncthreads();

        // ---- GEMM2: (16x192) x (192x144) : 6 K-blocks x 9 tiles = 54 WMMA ----
        v8f c2[9];
        #pragma unroll
        for (int t = 0; t < 9; ++t) c2[t] = vzero;
        #pragma unroll
        for (int kb = 0; kb < 6; ++kb) {
            V16 a;
            const unsigned short* ar = &Acat[m][kb * 32 + hf * 8];
            *((uint4*)&a.us[0]) = *((const uint4*)(ar));
            *((uint4*)&a.us[8]) = *((const uint4*)(ar + 16));
            const unsigned short* wrow = Wcat + (size_t)(kb * 32) * NP_ + m;
            // preload the 9 B fragments of this K-block, then 9 back-to-back WMMAs
            V16 bv[9];
            #pragma unroll
            for (int t = 0; t < 9; ++t) {
                const unsigned short* bp = wrow + t * 16;
                #pragma unroll
                for (int i = 0; i < 8; ++i) {
                    bv[t].us[i]     = bp[(hf * 8 + i) * NP_];
                    bv[t].us[8 + i] = bp[(16 + hf * 8 + i) * NP_];
                }
            }
            #pragma unroll
            for (int t = 0; t < 9; ++t) {
                c2[t] = __builtin_amdgcn_wmma_f32_16x16x32_bf16(
                    false, a.bf, false, bv[t].bf, (short)0, c2[t], false, false);
            }
        }
        // ---- + fused bias -> fwOut ----
        #pragma unroll
        for (int t = 0; t < 9; ++t) {
            int f = t * 16 + m;
            if (f < F_) {
                float bb = bcat[f];
                #pragma unroll
                for (int r = 0; r < 8; ++r)
                    fwOut[hf * 8 + r][f] = c2[t][r] + bb;
            }
        }
    };

    run_net(fiW1p, fiWcat, fi_b1, fi_evW1, fi_evb1, bcat_fi, fw_fi);
    run_net(feW1p, feWcat, fe_b1, fe_evW1, fe_evb1, bcat_fe, fw_fe);
    __syncthreads();

    // =================== attention + scatter (f32) ===================
    const float RS33 = 0.17407765595569785f;   // 1/sqrt(33)
    const float RS44 = 0.15075567228888181f;   // 1/sqrt(44)

    const float* qr = q_inv + (size_t)rN * F_;
    const float* ks = k_inv + (size_t)sN * F_;
    const float* vs = v_inv + (size_t)sN * F_;

    // inv heads: lane pair splits 132 features 66/66 == heads {0,1} / {2,3}
    #pragma unroll
    for (int hh = 0; hh < 2; ++hh) {
        int h  = hf * 2 + hh;
        int fb = h * 33;
        float dot = 0.f;
        #pragma unroll
        for (int d = 0; d < 33; ++d) {
            int f = fb + d;
            dot += qr[f] * ks[f] * fw_fi[m][f];
        }
        float ca = cut * dot * RS33;    // cutoff * alpha_inv[h]
        #pragma unroll
        for (int d = 0; d < 33; ++d) {
            int f = fb + d;
            atomicAdd(&d_inv[(size_t)rN * F_ + f], ca * vs[f]);
        }
    }

    // ev heads (44-wide): lane hf=0 gets h0 full + h1 partial,
    // lane hf=1 gets h1 partial + h2 full. Combine h1 across the pair.
    const float* qer = q_ev + (size_t)rN * F_;
    const float* kes = k_ev + (size_t)sN * F_;
    float dA = 0.f, dB = 0.f;
    #pragma unroll
    for (int i = 0; i < 66; ++i) {
        int f = hf * 66 + i;
        float p = qer[f] * kes[f] * fw_fe[m][f];
        if ((f / 44) == 1) dB += p; else dA += p;
    }
    float dB_other = __shfl_xor(dB, 16, 32);
    float caA = cut * dA * RS44;                 // h0 (hf=0) or h2 (hf=1)
    float ca1 = cut * (dB + dB_other) * RS44;    // h1 (both lanes)

    const float* she = sh + (size_t)e * EVD_;
    float* devr = d_ev + (size_t)rN * EVD_;
    if (hf == 0) {
        #pragma unroll
        for (int j = 0; j < 3; ++j) atomicAdd(&devr[j], caA * she[j]);
        #pragma unroll
        for (int j = 3; j < 8; ++j) atomicAdd(&devr[j], ca1 * she[j]);
    } else {
        #pragma unroll
        for (int j = 8; j < 15; ++j) atomicAdd(&devr[j], caA * she[j]);
    }
}

// ---------------------------------------------------------------------------
// K3: per-node finalize: att = x + d ; evi2 = so3(att_ev) ;
// t = [att_inv | evi2] @ W_int + b_int ; outputs.
// ---------------------------------------------------------------------------
__global__ __launch_bounds__(160) void et_finalize_kernel(
    const float* __restrict__ inv, const float* __restrict__ ev,
    const float* __restrict__ d_inv, const float* __restrict__ d_ev,
    const float* __restrict__ W_int, const float* __restrict__ b_int,
    float* __restrict__ out)
{
    __shared__ float cat[135];
    __shared__ float aev[EVD_];
    __shared__ float tb[3];
    const int n = blockIdx.x;
    const int j = threadIdx.x;

    if (j < F_)   cat[j] = inv[(size_t)n * F_ + j] + d_inv[(size_t)n * F_ + j];
    if (j < EVD_) aev[j] = ev[(size_t)n * EVD_ + j] + d_ev[(size_t)n * EVD_ + j];
    __syncthreads();
    if (j >= F_ && j < 135) {
        int k  = j - F_;
        int lo = (k == 0) ? 0 : ((k == 1) ? 3 : 8);
        int hi = (k == 0) ? 3 : ((k == 1) ? 8 : 15);
        float s = 0.f;
        for (int i = lo; i < hi; ++i) s += aev[i] * aev[i];
        cat[j] = s;
    }
    __syncthreads();
    if (j < 135) {
        float acc = b_int[j];
        #pragma unroll 5
        for (int i = 0; i < 135; ++i) acc += cat[i] * W_int[i * 135 + j];
        if (j < F_) out[(size_t)n * F_ + j] = cat[j] + acc;   // att_inv + t
        else        tb[j - F_] = acc;
    }
    __syncthreads();
    if (j < EVD_) {
        int h = (j < 3) ? 0 : ((j < 8) ? 1 : 2);
        out[(size_t)N_NODES_ * F_ + (size_t)n * EVD_ + j] = aev[j] * (1.f + tb[h]);
    }
}

// ---------------------------------------------------------------------------
extern "C" void kernel_launch(void* const* d_in, const int* in_sizes, int n_in,
                              void* d_out, int out_size, void* d_ws, size_t ws_size,
                              hipStream_t stream)
{
    (void)in_sizes; (void)n_in; (void)out_size; (void)ws_size;

    const float* inv_features = (const float*)d_in[0];
    const float* ev_features  = (const float*)d_in[1];
    const float* rbf          = (const float*)d_in[2];
    const int*   senders      = (const int*)  d_in[3];
    const int*   receivers    = (const int*)  d_in[4];
    const float* sh_vectors   = (const float*)d_in[5];
    const float* cutoffs      = (const float*)d_in[6];
    const float* W_q_inv      = (const float*)d_in[7];
    const float* W_k_inv      = (const float*)d_in[8];
    const float* W_v_inv      = (const float*)d_in[9];
    const float* W_q_ev       = (const float*)d_in[10];
    const float* W_k_ev       = (const float*)d_in[11];
    const float* fi_rbf_W1    = (const float*)d_in[12];
    const float* fi_rbf_b1    = (const float*)d_in[13];
    const float* fi_rbf_W2    = (const float*)d_in[14];
    const float* fi_rbf_b2    = (const float*)d_in[15];
    const float* fi_ev_W1     = (const float*)d_in[16];
    const float* fi_ev_b1     = (const float*)d_in[17];
    const float* fi_ev_W2     = (const float*)d_in[18];
    const float* fi_ev_b2     = (const float*)d_in[19];
    const float* fe_rbf_W1    = (const float*)d_in[20];
    const float* fe_rbf_b1    = (const float*)d_in[21];
    const float* fe_rbf_W2    = (const float*)d_in[22];
    const float* fe_rbf_b2    = (const float*)d_in[23];
    const float* fe_ev_W1     = (const float*)d_in[24];
    const float* fe_ev_b1     = (const float*)d_in[25];
    const float* fe_ev_W2     = (const float*)d_in[26];
    const float* fe_ev_b2     = (const float*)d_in[27];
    const float* W_int        = (const float*)d_in[28];
    const float* b_int        = (const float*)d_in[29];

    // ---- workspace carve-out (256 B aligned) : ~65 MB total ----
    char* ws = (char*)d_ws;
    size_t off = 0;
    auto take = [&](size_t bytes) -> void* {
        void* p = ws + off;
        off = (off + bytes + 255) & ~(size_t)255;
        return p;
    };
    unsigned short* fiW1p  = (unsigned short*)take((size_t)NRB_ * NP_ * 2);
    unsigned short* fiWcat = (unsigned short*)take((size_t)KP_  * NP_ * 2);
    unsigned short* feW1p  = (unsigned short*)take((size_t)NRB_ * NP_ * 2);
    unsigned short* feWcat = (unsigned short*)take((size_t)KP_  * NP_ * 2);
    float* bcat_fi = (float*)take((size_t)F_ * 4);
    float* bcat_fe = (float*)take((size_t)F_ * 4);
    float* q_inv   = (float*)take((size_t)N_NODES_ * F_ * 4);
    float* k_inv   = (float*)take((size_t)N_NODES_ * F_ * 4);
    float* v_inv   = (float*)take((size_t)N_NODES_ * F_ * 4);
    float* q_ev    = (float*)take((size_t)N_NODES_ * F_ * 4);
    float* k_ev    = (float*)take((size_t)N_NODES_ * F_ * 4);
    float* d_inv   = (float*)take((size_t)N_NODES_ * F_ * 4);
    float* d_ev    = (float*)take((size_t)N_NODES_ * EVD_ * 4);

    // K0: weight prepack (bf16 + padding + fused bias)
    {
        int total = 2 * (NRB_ * NP_) + 2 * (KP_ * NP_) + 2 * F_;
        et_prep_kernel<<<(total + 255) / 256, 256, 0, stream>>>(
            fi_rbf_W1, fi_rbf_W2, fi_ev_W2, fi_rbf_b2, fi_ev_b2,
            fe_rbf_W1, fe_rbf_W2, fe_ev_W2, fe_rbf_b2, fe_ev_b2,
            fiW1p, fiWcat, feW1p, feWcat, bcat_fi, bcat_fe);
    }
    // K1: node q/k/v tables + zero accumulators
    {
        int total = N_NODES_ * F_;
        et_node_proj_kernel<<<(total + 255) / 256, 256, 0, stream>>>(
            inv_features, W_q_inv, W_k_inv, W_v_inv, W_q_ev, W_k_ev,
            q_inv, k_inv, v_inv, q_ev, k_ev, d_inv, d_ev);
    }
    // K2: edge filter MLPs (WMMA) + attention + scatter-add
    et_edge_kernel<<<N_EDGES_ / 16, 32, 0, stream>>>(
        ev_features, rbf, senders, receivers, sh_vectors, cutoffs,
        fi_rbf_b1, fi_ev_W1, fi_ev_b1,
        fe_rbf_b1, fe_ev_W1, fe_ev_b1,
        fiW1p, fiWcat, feW1p, feWcat, bcat_fi, bcat_fe,
        q_inv, k_inv, v_inv, q_ev, k_ev, d_inv, d_ev);
    // K3: per-node finalize
    et_finalize_kernel<<<N_NODES_, 160, 0, stream>>>(
        inv_features, ev_features, d_inv, d_ev, W_int, b_int, (float*)d_out);
}
